// EdgeConv_67800353734678
// MI455X (gfx1250) — matrix operand order repaired and checked by reading, not compiled, not run on hardware
//
#include <hip/hip_runtime.h>

typedef __attribute__((ext_vector_type(16))) __bf16   v16bf;
typedef __attribute__((ext_vector_type(8)))  float    v8f;
typedef __attribute__((ext_vector_type(4)))  unsigned u32x4;
typedef __attribute__((ext_vector_type(4)))  int      i32x4;
typedef __attribute__((ext_vector_type(4)))  float    f32x4;

#define BATCH 256
#define CDIM  256
#define LDIM  512
#define MDIM  512   // stacked [W1;W2] rows
#define KDIM  256   // reduction = C
#define MBLK  128
#define NBLK  128
#define KSTEP 64
#define KP    72    // padded LDS K-stride (halfs): 36 dwords -> conflict-free b128 frag loads

#if defined(__has_builtin)
#if __has_builtin(__builtin_amdgcn_global_load_async_to_lds_b128)
#define HAVE_ASYNC_LDS 1
#endif
#if __has_builtin(__builtin_amdgcn_s_wait_asynccnt)
#define HAVE_WAIT_ASYNC 1
#endif
#endif

__device__ __forceinline__ unsigned short f2bf(float f) {
    unsigned u = __float_as_uint(f);
    unsigned r = u + 0x7FFFu + ((u >> 16) & 1u);   // round-to-nearest-even
    return (unsigned short)(r >> 16);
}
__device__ __forceinline__ float bf2f(unsigned short h) {
    return __uint_as_float(((unsigned)h) << 16);
}

union Frag16 { u32x4 q[2]; v16bf v; };

// ---- Pass 0: W [256][512] fp32 -> Wstack [512][256] bf16 (W1 on top, W2 below)
__global__ __launch_bounds__(256) void convw_kernel(const float* __restrict__ W,
                                                    unsigned short* __restrict__ Wb) {
    int g = blockIdx.x * 256 + threadIdx.x;        // < 512*256
    int m = g >> 8;
    int k = g & 255;
    float v = (m < CDIM) ? W[m * (2 * CDIM) + k]
                         : W[(m - CDIM) * (2 * CDIM) + CDIM + k];
    Wb[g] = f2bf(v);
}

// ---- Pass 1: per batch  O(512x512) = Wstack(512x256,bf16) @ x[b](256x512 -> bf16)
// rows <256 -> Y1 (fp32, into d_out), rows >=256 -> Z (bf16, into ws)
__global__ __launch_bounds__(256) void gemm_kernel(const float* __restrict__ x,
                                                   const unsigned short* __restrict__ Wb,
                                                   float* __restrict__ y1,
                                                   unsigned short* __restrict__ z) {
    __shared__ unsigned short Asm[MBLK * KP];   // [m][k]  (k contiguous)
    __shared__ unsigned short Bsm[NBLK * KP];   // [n][k]  (transposed store, k contiguous)

    const int b  = blockIdx.z;
    const int m0 = blockIdx.y * MBLK;
    const int n0 = blockIdx.x * NBLK;
    const int t  = threadIdx.x;
    const int wid  = t >> 5;
    const int lane = t & 31;
    const int wm = wid >> 2;          // 0..1  (64-row stripe)
    const int wn = wid & 3;           // 0..3  (32-col stripe)
    const int lr = lane & 15;
    const int hi = lane >> 4;

    const float* xb = x + (size_t)b * KDIM * LDIM;

    v8f acc[4][2];
    v8f zero = {0.f, 0.f, 0.f, 0.f, 0.f, 0.f, 0.f, 0.f};
#pragma unroll
    for (int tm = 0; tm < 4; ++tm)
#pragma unroll
        for (int tn = 0; tn < 2; ++tn) acc[tm][tn] = zero;

    for (int ks = 0; ks < KDIM; ks += KSTEP) {
        __syncthreads();
        // Stage A: Wb rows m0..m0+127, k ks..ks+63.
        // Pure bf16 copy -> CDNA5 async global->LDS engine (ASYNCcnt),
        // overlapping with the register-staged convert/transpose of B below.
#if defined(HAVE_ASYNC_LDS)
        for (int i = t; i < (MBLK * KSTEP / 8); i += 256) {   // 1024 b128 chunks
            int row = i >> 3;
            int k8  = (i & 7) << 3;
            __builtin_amdgcn_global_load_async_to_lds_b128(
                (i32x4*)&Wb[(m0 + row) * KDIM + ks + k8],
                (i32x4*)&Asm[row * KP + k8],
                0, 0);
        }
#else
        for (int i = t; i < (MBLK * KSTEP / 8); i += 256) {   // 1024 uint4
            int row = i >> 3;
            int k8  = (i & 7) << 3;
            *(u32x4*)&Asm[row * KP + k8] =
                *(const u32x4*)&Wb[(m0 + row) * KDIM + ks + k8];
        }
#endif
        // Stage B: x[b][ks..ks+63][n0..n0+127] fp32 -> bf16, transposed to [n][k]
        for (int i = t; i < (KSTEP * NBLK / 4); i += 256) {   // 2048 float4
            int c  = i >> 5;
            int l4 = (i & 31) << 2;
            f32x4 v = *(const f32x4*)&xb[(size_t)(ks + c) * LDIM + n0 + l4];
            Bsm[(l4 + 0) * KP + c] = f2bf(v.x);
            Bsm[(l4 + 1) * KP + c] = f2bf(v.y);
            Bsm[(l4 + 2) * KP + c] = f2bf(v.z);
            Bsm[(l4 + 3) * KP + c] = f2bf(v.w);
        }
        if (ks + KSTEP < KDIM) {   // hint next K-chunk toward near caches
            __builtin_prefetch(&Wb[(m0 + (t >> 1)) * KDIM + ks + KSTEP], 0, 3);
            __builtin_prefetch(&xb[(size_t)(ks + KSTEP + (t >> 2)) * LDIM + n0], 0, 3);
        }
#if defined(HAVE_ASYNC_LDS)
#if defined(HAVE_WAIT_ASYNC)
        __builtin_amdgcn_s_wait_asynccnt(0);
#else
        asm volatile("s_wait_asynccnt 0x0" ::: "memory");
#endif
#endif
        __syncthreads();

#pragma unroll
        for (int kk = 0; kk < KSTEP; kk += 32) {
            Frag16 af[4];
#pragma unroll
            for (int tm = 0; tm < 4; ++tm) {
                // 16-bit A layout: lane<16 holds K {kk..kk+7, kk+16..kk+23} of row m
                const unsigned short* p =
                    &Asm[(wm * 64 + tm * 16 + lr) * KP + kk + hi * 8];
                af[tm].q[0] = *(const u32x4*)p;
                af[tm].q[1] = *(const u32x4*)(p + 16);
            }
            Frag16 bfr[2];
#pragma unroll
            for (int tn = 0; tn < 2; ++tn) {
                // 16-bit B layout: lane<16 holds K {kk..kk+7, kk+16..kk+23} of column n
                const unsigned short* p =
                    &Bsm[(wn * 32 + tn * 16 + lr) * KP + kk + hi * 8];
                bfr[tn].q[0] = *(const u32x4*)p;
                bfr[tn].q[1] = *(const u32x4*)(p + 16);
            }
#pragma unroll
            for (int tm = 0; tm < 4; ++tm)
#pragma unroll
                for (int tn = 0; tn < 2; ++tn)
                    acc[tm][tn] = __builtin_amdgcn_wmma_f32_16x16x32_bf16(
                        false, af[tm].v, false, bfr[tn].v,
                        (short)0, acc[tm][tn], false, false);
        }
    }

    // Store: C/D layout -> lane (lr,hi) vgpr v = element (m = hi*8+v, n = lr)
    const bool isZ = (m0 >= CDIM);
#pragma unroll
    for (int tm = 0; tm < 4; ++tm) {
#pragma unroll
        for (int tn = 0; tn < 2; ++tn) {
            int mg = m0 + wm * 64 + tm * 16 + hi * 8;
            int ng = n0 + wn * 32 + tn * 16 + lr;
            v8f a = acc[tm][tn];
            if (!isZ) {
                float* dst = y1 + ((size_t)(b * CDIM + mg)) * LDIM + ng;
#pragma unroll
                for (int v = 0; v < 8; ++v) dst[(size_t)v * LDIM] = a[v];
            } else {
                unsigned short* dst =
                    z + ((size_t)(b * CDIM + (mg - CDIM))) * LDIM + ng;
#pragma unroll
                for (int v = 0; v < 8; ++v) dst[(size_t)v * LDIM] = f2bf(a[v]);
            }
        }
    }
}

// ---- Pass 2: y = Y1 + gather_L(Z) ; InstanceNorm over L ; ReLU  (in place on d_out)
// Bias cancels under mean subtraction, so it is never applied.
__global__ __launch_bounds__(256) void inorm_kernel(const int* __restrict__ idx,
                                                    const unsigned short* __restrict__ z,
                                                    float* __restrict__ y) {
    __shared__ int idx_s[LDIM];
    __shared__ unsigned short zrow[8][520];     // per-wave row; slot 512 = zero pad

    const int b    = blockIdx.x;
    const int t    = threadIdx.x;
    const int wid  = t >> 5;
    const int lane = t & 31;

    idx_s[t]       = idx[b * LDIM + t];
    idx_s[t + 256] = idx[b * LDIM + 256 + t];
    __syncthreads();

    for (int it = 0; it < 32; ++it) {
        int o = it * 8 + wid;
        const unsigned* zr32 = (const unsigned*)(z + ((size_t)(b * CDIM + o)) * LDIM);
        unsigned* zb32 = (unsigned*)&zrow[wid][0];
#pragma unroll
        for (int j = 0; j < 8; ++j) zb32[lane + 32 * j] = zr32[lane + 32 * j];
        if (lane == 0) zrow[wid][512] = 0;
        __syncthreads();

        float* yr = y + ((size_t)(b * CDIM + o)) * LDIM;
        float vals[16];
        float s = 0.f, s2 = 0.f;
#pragma unroll
        for (int j = 0; j < 16; ++j) {
            int l = lane + 32 * j;
            float v = yr[l] + bf2f(zrow[wid][idx_s[l]]);
            vals[j] = v;
            s += v;
            s2 += v * v;
        }
#pragma unroll
        for (int off = 16; off > 0; off >>= 1) {
            s  += __shfl_xor(s, off, 32);
            s2 += __shfl_xor(s2, off, 32);
        }
        float mean = s * (1.0f / (float)LDIM);
        float var  = s2 * (1.0f / (float)LDIM) - mean * mean;
        float rs   = rsqrtf(var + 1e-5f);
#pragma unroll
        for (int j = 0; j < 16; ++j) {
            float v = (vals[j] - mean) * rs;
            yr[lane + 32 * j] = v > 0.f ? v : 0.f;
        }
        __syncthreads();
    }
}

extern "C" void kernel_launch(void* const* d_in, const int* in_sizes, int n_in,
                              void* d_out, int out_size, void* d_ws, size_t ws_size,
                              hipStream_t stream) {
    const float* x   = (const float*)d_in[0];
    const int*   idx = (const int*)d_in[1];
    const float* W   = (const float*)d_in[2];
    // d_in[3] = bias: provably cancelled by InstanceNorm mean subtraction.

    unsigned short* Wb = (unsigned short*)d_ws;                       // 512*256*2 = 256 KiB
    unsigned short* Z  = (unsigned short*)((char*)d_ws + MDIM * KDIM * 2);  // 64 MiB bf16
    float* out = (float*)d_out;

    convw_kernel<<<dim3(512), dim3(256), 0, stream>>>(W, Wb);

    dim3 g1(LDIM / NBLK, MDIM / MBLK, BATCH);   // (4, 4, 256)
    gemm_kernel<<<g1, dim3(256), 0, stream>>>(x, Wb, out, Z);

    inorm_kernel<<<dim3(BATCH), dim3(256), 0, stream>>>(idx, Z, out);
}
